// OneEikonalStep_5789615915253
// MI455X (gfx1250) — compile-verified, matrix operand order checked by reading, not run on hardware
//
#include <hip/hip_runtime.h>
#include <hip/hip_bf16.h>

typedef __attribute__((ext_vector_type(16))) _Float16 v16h;
typedef __attribute__((ext_vector_type(8)))  float    v8f;

union HFrag { v16h v; _Float16 h[16]; uint4 q[2]; };

// A-fragment (16x32 f16 tile, rows striped per lane) from an LDS activation
// pane with 64-half row stride. Per CDNA5 ISA layout: lanes 0-15 hold
// K = koff+{0..7, 16..23}, lanes 16-31 hold K = koff+{8..15, 24..31}.
// Both 8-half runs are contiguous & 16B aligned -> two b128 LDS loads.
__device__ __forceinline__ v16h load_afrag(const _Float16* act, int row, int koff, int lane) {
  HFrag u;
  const int kb = koff + ((lane & 16) ? 8 : 0);
  const uint4* p = (const uint4*)(act + row * 64 + kb);
  u.q[0] = p[0];   // halves kb .. kb+7
  u.q[1] = p[2];   // halves kb+16 .. kb+23
  return u.v;
}

// B-fragment (32x16 f16, cols striped per lane) from LDS weight matrix with
// row stride `stride` halves. Done once at kernel start (weights reused 32x).
__device__ __forceinline__ v16h load_bfrag(const _Float16* W, int stride, int colBase,
                                           int koff, int lane) {
  HFrag u;
  const int c  = colBase + (lane & 15);
  const int kb = koff + ((lane & 16) ? 8 : 0);
#pragma unroll
  for (int j = 0; j < 8; ++j) {
    u.h[j]     = W[(kb + j) * stride + c];
    u.h[j + 8] = W[(kb + 16 + j) * stride + c];
  }
  return u.v;
}

__device__ __forceinline__ v8f wmma_f16(v16h a, v16h b, v8f c) {
  return __builtin_amdgcn_wmma_f32_16x16x32_f16(false, a, false, b, (short)0, c, false, false);
}

__global__ __launch_bounds__(256)
void eikonal_march_kernel(const float* __restrict__ rp0, const float* __restrict__ rd0,
                          const float* __restrict__ rt0, const float* __restrict__ grid,
                          const float* __restrict__ W1, const float* __restrict__ b1,
                          const float* __restrict__ W2, const float* __restrict__ b2,
                          const float* __restrict__ W3, const float* __restrict__ b3,
                          const int* __restrict__ nsteps_p,
                          float* __restrict__ out, int N)
{
  __shared__ _Float16 W1h[32 * 64];                 // 27x64 zero-padded to 32x64
  __shared__ _Float16 W2h[64 * 64];
  __shared__ _Float16 W3h[64 * 16];                 // 64x3 zero-padded to 64x16
  __shared__ float b1s[64];
  __shared__ float b2s[64];
  __shared__ float b3s[4];
  __shared__ __align__(16) _Float16 acts[8][32 * 64]; // per-wave activation pane (4KB)

  const int tid    = threadIdx.x;
  const int lane   = tid & 31;
  const int wave   = tid >> 5;
  const int ray_g  = blockIdx.x * 256 + tid;
  const int ray    = (ray_g < N) ? ray_g : (N - 1);
  const int nsteps = *nsteps_p;

  // ---- stage weights f32 -> f16 into LDS (zero padded) ----
  for (int i = tid; i < 32 * 64; i += 256) {
    int r = i >> 6, c = i & 63;
    W1h[i] = (r < 27) ? (_Float16)W1[r * 64 + c] : (_Float16)0.f;
  }
  for (int i = tid; i < 64 * 64; i += 256) W2h[i] = (_Float16)W2[i];
  for (int i = tid; i < 64 * 16; i += 256) {
    int r = i >> 4, c = i & 15;
    W3h[i] = (c < 3) ? (_Float16)W3[r * 3 + c] : (_Float16)0.f;
  }
  if (tid < 64) { b1s[tid] = b1[tid]; b2s[tid] = b2[tid]; }
  if (tid < 4)  b3s[tid] = (tid < 3) ? b3[tid] : 0.f;
  __syncthreads();

  // ---- B fragments resident in VGPRs for the whole march ----
  v16h B1[4], B2lo[4], B2hi[4], B3lo, B3hi;
#pragma unroll
  for (int n = 0; n < 4; ++n) {
    B1[n]   = load_bfrag(W1h, 64, n * 16, 0,  lane);
    B2lo[n] = load_bfrag(W2h, 64, n * 16, 0,  lane);
    B2hi[n] = load_bfrag(W2h, 64, n * 16, 32, lane);
  }
  B3lo = load_bfrag(W3h, 16, 0, 0,  lane);
  B3hi = load_bfrag(W3h, 16, 0, 32, lane);

  _Float16* act  = acts[wave];
  float*    predf = (float*)acts[wave];   // overlay for layer-3 results

  // ---- ray state in registers across all steps ----
  float px = rp0[ray * 3 + 0], py = rp0[ray * 3 + 1], pz = rp0[ray * 3 + 2];
  float dx = rd0[ray * 3 + 0], dy = rd0[ray * 3 + 1], dz = rd0[ray * 3 + 2];
  float rt = rt0[ray];

  const int halfsel = (lane & 16) ? 8 : 0;  // D-tile row offset for hi lanes
  const int mrow    = lane & 15;
  const int mycol   = lane & 15;
  const float umax  = 127.f - 1e-4f;

  for (int s = 0; s < nsteps; ++s) {
    // ===== trilerp value + analytic gradient (grid L2-resident, 8MB) =====
    float urx = (px + 1.f) * 63.5f;
    float ury = (py + 1.f) * 63.5f;
    float urz = (pz + 1.f) * 63.5f;
    float ux = fminf(fmaxf(urx, 0.f), umax);
    float uy = fminf(fmaxf(ury, 0.f), umax);
    float uz = fminf(fmaxf(urz, 0.f), umax);
    int ix = (int)floorf(ux), iy = (int)floorf(uy), iz = (int)floorf(uz);
    float fx = ux - (float)ix, fy = uy - (float)iy, fz = uz - (float)iz;
    const float* gp = grid + (((size_t)ix * 128 + (size_t)iy) * 128 + (size_t)iz);
    float c000 = gp[0],             c001 = gp[1];
    float c010 = gp[128],           c011 = gp[129];
    float c100 = gp[16384],         c101 = gp[16385];
    float c110 = gp[16512],         c111 = gp[16513];
    float c00 = c000 + (c100 - c000) * fx;
    float c10 = c010 + (c110 - c010) * fx;
    float c01 = c001 + (c101 - c001) * fx;
    float c11 = c011 + (c111 - c011) * fx;
    float c0  = c00 + (c10 - c00) * fy;
    float c1  = c01 + (c11 - c01) * fy;
    float val = c0 + (c1 - c0) * fz;
    float dfx0 = (c100 - c000) + ((c110 - c010) - (c100 - c000)) * fy;
    float dfx1 = (c101 - c001) + ((c111 - c011) - (c101 - c001)) * fy;
    float dfx  = dfx0 + (dfx1 - dfx0) * fz;
    float dfy  = (c10 - c00) + ((c11 - c01) - (c10 - c00)) * fz;
    float dfz  = c1 - c0;
    float gx = dfx * ((urx > 0.f && urx < umax) ? 63.5f : 0.f);
    float gy = dfy * ((ury > 0.f && ury < umax) ? 63.5f : 0.f);
    float gz = dfz * ((urz > 0.f && urz < umax) ? 63.5f : 0.f);

    // ===== posenc (alpha fully annealed -> window weights == 1) =====
    float feats[27];
    feats[0] = px; feats[1] = py; feats[2] = pz;
#pragma unroll
    for (int j = 0; j < 4; ++j) {
      float fq = (float)(1 << j);
      feats[3 + j * 6 + 0] = __sinf(px * fq);
      feats[3 + j * 6 + 1] = __sinf(py * fq);
      feats[3 + j * 6 + 2] = __sinf(pz * fq);
      feats[3 + j * 6 + 3] = __cosf(px * fq);
      feats[3 + j * 6 + 4] = __cosf(py * fq);
      feats[3 + j * 6 + 5] = __cosf(pz * fq);
    }
    _Float16* frow = act + lane * 64;
#pragma unroll
    for (int k = 0; k < 27; ++k) frow[k] = (_Float16)feats[k];
#pragma unroll
    for (int k = 27; k < 32; ++k) frow[k] = (_Float16)0.f;
    __syncthreads();

    // ===== layer 1: [32x32] x [32x64], K=32 -> 1 WMMA per (tile, ntile) =====
    v16h a1_0 = load_afrag(act, mrow,      0, lane);
    v16h a1_1 = load_afrag(act, 16 + mrow, 0, lane);
    __syncthreads();
#pragma unroll
    for (int n = 0; n < 4; ++n) {
      float bias = b1s[n * 16 + mycol];
      v8f acc0 = {}; acc0 = wmma_f16(a1_0, B1[n], acc0);
      v8f acc1 = {}; acc1 = wmma_f16(a1_1, B1[n], acc1);
      _Float16* hc = act + n * 16 + mycol;
#pragma unroll
      for (int v = 0; v < 8; ++v) {
        hc[(v + halfsel) * 64]      = (_Float16)fmaxf(acc0[v] + bias, 0.f);
        hc[(16 + v + halfsel) * 64] = (_Float16)fmaxf(acc1[v] + bias, 0.f);
      }
    }
    __syncthreads();

    // ===== layer 2: [32x64] x [64x64], K=64 -> 2 chained WMMA =====
    v16h a2l0 = load_afrag(act, mrow,      0,  lane);
    v16h a2h0 = load_afrag(act, mrow,      32, lane);
    v16h a2l1 = load_afrag(act, 16 + mrow, 0,  lane);
    v16h a2h1 = load_afrag(act, 16 + mrow, 32, lane);
    __syncthreads();
#pragma unroll
    for (int n = 0; n < 4; ++n) {
      float bias = b2s[n * 16 + mycol];
      v8f acc0 = {}; acc0 = wmma_f16(a2l0, B2lo[n], acc0); acc0 = wmma_f16(a2h0, B2hi[n], acc0);
      v8f acc1 = {}; acc1 = wmma_f16(a2l1, B2lo[n], acc1); acc1 = wmma_f16(a2h1, B2hi[n], acc1);
      _Float16* hc = act + n * 16 + mycol;
#pragma unroll
      for (int v = 0; v < 8; ++v) {
        hc[(v + halfsel) * 64]      = (_Float16)fmaxf(acc0[v] + bias, 0.f);
        hc[(16 + v + halfsel) * 64] = (_Float16)fmaxf(acc1[v] + bias, 0.f);
      }
    }
    __syncthreads();

    // ===== layer 3: [32x64] x [64x16(pad of 3)] =====
    v16h a3l0 = load_afrag(act, mrow,      0,  lane);
    v16h a3h0 = load_afrag(act, mrow,      32, lane);
    v16h a3l1 = load_afrag(act, 16 + mrow, 0,  lane);
    v16h a3h1 = load_afrag(act, 16 + mrow, 32, lane);
    __syncthreads();
    v8f p0 = {}; p0 = wmma_f16(a3l0, B3lo, p0); p0 = wmma_f16(a3h0, B3hi, p0);
    v8f p1 = {}; p1 = wmma_f16(a3l1, B3lo, p1); p1 = wmma_f16(a3h1, B3hi, p1);
    if (mycol < 3) {
      float bias = b3s[mycol];
#pragma unroll
      for (int v = 0; v < 8; ++v) {
        predf[(v + halfsel) * 4 + mycol]        = p0[v] + bias;
        predf[(16 + v + halfsel) * 4 + mycol]   = p1[v] + bias;
      }
    }
    __syncthreads();

    // ===== state update + outputs =====
    float pgx = predf[lane * 4 + 0];
    float pgy = predf[lane * 4 + 1];
    float pgz = predf[lane * 4 + 2];
    bool  sel = sqrtf(gx * gx + gy * gy + gz * gz) > 0.001f;
    float ggx = sel ? pgx : gx;
    float ggy = sel ? pgy : gy;
    float ggz = sel ? pgz : gz;
    float stepw = 0.01f / val;
    float npx = px + stepw * dx, npy = py + stepw * dy, npz = pz + stepw * dz;
    float ndx = dx + 0.01f * ggx, ndy = dy + 0.01f * ggy, ndz = dz + 0.01f * ggz;
    float ex = px - npx, ey = py - npy, ez = pz - npz;
    float nrt = rt + sqrtf(ex * ex + ey * ey + ez * ez);

    if (ray_g < N) {
      float* o = out + ((size_t)s * (size_t)N + (size_t)ray_g) * 11;
      o[0] = npx; o[1] = npy; o[2] = npz;
      o[3] = ndx; o[4] = ndy; o[5] = ndz;
      o[6] = nrt; o[7] = val;
      o[8] = gx;  o[9] = gy;  o[10] = gz;
    }
    px = npx; py = npy; pz = npz;
    dx = ndx; dy = ndy; dz = ndz;
    rt = nrt;
    __syncthreads();   // pred pane is recycled as feature pane next step
  }
}

extern "C" void kernel_launch(void* const* d_in, const int* in_sizes, int n_in,
                              void* d_out, int out_size, void* d_ws, size_t ws_size,
                              hipStream_t stream) {
  const float* rp0  = (const float*)d_in[0];
  const float* rd0  = (const float*)d_in[1];
  const float* rt0  = (const float*)d_in[2];
  const float* grid = (const float*)d_in[3];
  const float* W1   = (const float*)d_in[4];
  const float* b1   = (const float*)d_in[5];
  const float* W2   = (const float*)d_in[6];
  const float* b2   = (const float*)d_in[7];
  const float* W3   = (const float*)d_in[8];
  const float* b3   = (const float*)d_in[9];
  const int*   nst  = (const int*)d_in[10];
  const int N = in_sizes[2];                 // rt0 is [N,1]
  const int blocks = (N + 255) / 256;
  eikonal_march_kernel<<<blocks, 256, 0, stream>>>(
      rp0, rd0, rt0, grid, W1, b1, W2, b2, W3, b3, nst, (float*)d_out, N);
}
